// MultiCropContrastive_Loss_v2_46334107189855
// MI455X (gfx1250) — matrix operand order repaired and checked by reading, not compiled, not run on hardware
//
#include <hip/hip_runtime.h>
#include <hip/hip_bf16.h>

typedef __attribute__((ext_vector_type(16))) __bf16 v16bf;
typedef __attribute__((ext_vector_type(8)))  __bf16 v8bf;
typedef __attribute__((ext_vector_type(8)))  float  v8f;

#define NP   16
#define BD   512
#define DIM  2048
#define PB   (NP * BD)      // 8192 rows of f
#define INV_TEMP 5.0f       // 1 / 0.2

// ---------------- kernel 1: zsum[b,d] = sum_p features[p,b,d]; also zero out ----------------
__global__ void zsum_kernel(const float* __restrict__ feat, float* __restrict__ zsum,
                            float* __restrict__ out) {
    if (blockIdx.x == 0 && threadIdx.x == 0) out[0] = 0.0f;
    int idx = blockIdx.x * 256 + threadIdx.x;          // over BD*DIM = 1M elements
    if (idx < BD * DIM) {
        float s = 0.0f;
        #pragma unroll
        for (int p = 0; p < NP; ++p)
            s += feat[(size_t)p * BD * DIM + idx];
        zsum[idx] = s;
    }
}

// ---------------- kernel 2: per-row L2-normalize + f32->bf16 in one pass ----------------
__global__ __launch_bounds__(256) void normalize_rows_kernel(
    const float* __restrict__ feat, const float* __restrict__ zsum,
    __bf16* __restrict__ fhat, __bf16* __restrict__ zhat)
{
    int row = blockIdx.x;
    const float* src;
    __bf16* dst;
    if (row < PB) { src = feat + (size_t)row * DIM;        dst = fhat + (size_t)row * DIM; }
    else          { src = zsum + (size_t)(row - PB) * DIM; dst = zhat + (size_t)(row - PB) * DIM; }

    const int tid = threadIdx.x;
    const float4* s4 = (const float4*)src;
    float4 x0 = s4[tid * 2];
    float4 x1 = s4[tid * 2 + 1];

    float s = x0.x * x0.x + x0.y * x0.y + x0.z * x0.z + x0.w * x0.w
            + x1.x * x1.x + x1.y * x1.y + x1.z * x1.z + x1.w * x1.w;
    #pragma unroll
    for (int off = 16; off > 0; off >>= 1)
        s += __shfl_xor(s, off, 32);

    __shared__ float wsum[8];
    __shared__ float inv_norm;
    if ((tid & 31) == 0) wsum[tid >> 5] = s;
    __syncthreads();
    if (tid == 0) {
        float t = 0.0f;
        #pragma unroll
        for (int i = 0; i < 8; ++i) t += wsum[i];
        inv_norm = 1.0f / fmaxf(sqrtf(t), 1e-12f);
    }
    __syncthreads();

    float inv = inv_norm;
    v8bf o;
    o[0] = (__bf16)(x0.x * inv); o[1] = (__bf16)(x0.y * inv);
    o[2] = (__bf16)(x0.z * inv); o[3] = (__bf16)(x0.w * inv);
    o[4] = (__bf16)(x1.x * inv); o[5] = (__bf16)(x1.y * inv);
    o[6] = (__bf16)(x1.z * inv); o[7] = (__bf16)(x1.w * inv);
    ((v8bf*)dst)[tid] = o;
}

// ---------------- kernel 3: bf16-WMMA GEMM + fused row logsumexp ----------------
__device__ __forceinline__ v16bf cat8(v8bf lo, v8bf hi) {
    union { v16bf v; v8bf h[2]; } u;
    u.h[0] = lo; u.h[1] = hi;
    return u.v;
}

__global__ __launch_bounds__(256) void gemm_lse_kernel(
    const __bf16* __restrict__ fhat,   // [PB, DIM]
    const __bf16* __restrict__ zhat,   // [BD, DIM]
    float* __restrict__ out)
{
    __shared__ __align__(16) __bf16 astage[2][16 * 128];  // 2 x 4 KB A staging
    __shared__ float sim[16 * BD];                        // 16 x 512 f32 = 32 KB

    const int tid  = threadIdx.x;
    const int wave = tid >> 5;
    const int lane = tid & 31;
    const int half = lane >> 4;
    const int l16  = lane & 15;
    const int m0   = blockIdx.x * 16;

    // async A staging: thread t copies 16B: row (t>>4), segment (t&15)
    const int arow_ld = tid >> 4;
    const int aseg    = tid & 15;
    const __bf16* agbase = fhat + (size_t)(m0 + arow_ld) * DIM + aseg * 8;
    const unsigned lds_dst0 = (unsigned)(size_t)&astage[0][arow_ld * 128 + aseg * 8];
    const unsigned lds_dst1 = (unsigned)(size_t)&astage[1][arow_ld * 128 + aseg * 8];

    // B: lane feeds z-row (wave*64 + t*16 + l16); K = 8*half+[0,8) and 16+8*half+[0,8)
    const __bf16* bptr[4];
    #pragma unroll
    for (int t = 0; t < 4; ++t)
        bptr[t] = zhat + (size_t)(wave * 64 + t * 16 + l16) * DIM;

    v8f acc[4] = {};

    auto loadB = [&](v16bf* d, int koff) {
        #pragma unroll
        for (int t = 0; t < 4; ++t)
            d[t] = cat8(*(const v8bf*)(bptr[t] + koff + 8 * half),
                        *(const v8bf*)(bptr[t] + koff + 16 + 8 * half));
    };
    auto loadA = [&](const __bf16* ab, int aoff) -> v16bf {
        return cat8(*(const v8bf*)(ab + aoff + 8 * half),
                    *(const v8bf*)(ab + aoff + 16 + 8 * half));
    };
    auto doWmma = [&](v16bf a, const v16bf* b) {
        #pragma unroll
        for (int t = 0; t < 4; ++t)
            acc[t] = __builtin_amdgcn_wmma_f32_16x16x32_bf16(
                false, a, false, b[t], (short)0, acc[t], false, false);
    };

    // one 128-K stage: preloaded A frags + ping-pong B pipeline, regions fenced
    // so loads cannot sink into the WMMA groups that consume them.
    auto stagebody = [&](int kb, const __bf16* ab, unsigned dstNext, bool issueNext) {
        if (issueNext) {
            unsigned long long ga = (unsigned long long)(size_t)(agbase + kb + 128);
            asm volatile("global_load_async_to_lds_b128 %0, %1, off"
                         :: "v"(dstNext), "v"(ga) : "memory");
        }
        v16bf bA[4], bB[4];
        loadB(bA, kb +  0);
        loadB(bB, kb + 32);
        v16bf a0 = loadA(ab,  0);
        v16bf a1 = loadA(ab, 32);
        v16bf a2 = loadA(ab, 64);
        v16bf a3 = loadA(ab, 96);
        __builtin_amdgcn_sched_barrier(0);
        doWmma(a0, bA);
        loadB(bA, kb + 64);
        __builtin_amdgcn_sched_barrier(0);
        doWmma(a1, bB);
        loadB(bB, kb + 96);
        __builtin_amdgcn_sched_barrier(0);
        doWmma(a2, bA);
        __builtin_amdgcn_sched_barrier(0);
        doWmma(a3, bB);
        if (issueNext) {
            asm volatile("s_wait_asynccnt 0x0" ::: "memory");
            __syncthreads();   // next stage staged for all waves; WAR-protects buffers
        }
    };

    // prologue: stage k=0 chunk of A into LDS buffer 0
    {
        unsigned long long ga = (unsigned long long)(size_t)agbase;
        asm volatile("global_load_async_to_lds_b128 %0, %1, off"
                     :: "v"(lds_dst0), "v"(ga) : "memory");
        asm volatile("s_wait_asynccnt 0x0" ::: "memory");
    }
    __syncthreads();

    const __bf16* ab0 = &astage[0][l16 * 128];
    const __bf16* ab1 = &astage[1][l16 * 128];
    for (int kb = 0; kb < DIM; kb += 256) {
        stagebody(kb,       ab0, lds_dst1, true);            // kb+128 < DIM always
        stagebody(kb + 128, ab1, lds_dst0, kb + 256 < DIM);
    }

    // Spill accumulators to LDS: VGPR i -> lanes 0-15 = (M=i, N=lane), lanes 16-31 = (M=i+8, N=lane-16)
    #pragma unroll
    for (int t = 0; t < 4; ++t) {
        int n = wave * 64 + t * 16 + l16;
        #pragma unroll
        for (int i = 0; i < 8; ++i) {
            int m = i + 8 * half;
            sim[m * BD + n] = acc[t][i];
        }
    }
    __syncthreads();

    // Per-row logsumexp over 512 cols; 16 rows -> 2 per wave.
    float partial = 0.0f;
    #pragma unroll
    for (int rr = 0; rr < 2; ++rr) {
        int r = wave * 2 + rr;
        const float* row = &sim[r * BD];

        float mx = -INFINITY;
        for (int c = lane; c < BD; c += 32) mx = fmaxf(mx, row[c]);
        #pragma unroll
        for (int off = 16; off > 0; off >>= 1)
            mx = fmaxf(mx, __shfl_xor(mx, off, 32));

        float se = 0.0f;
        for (int c = lane; c < BD; c += 32)
            se += __expf((row[c] - mx) * INV_TEMP);
        #pragma unroll
        for (int off = 16; off > 0; off >>= 1)
            se += __shfl_xor(se, off, 32);

        if (lane == 0) {
            int   grow = m0 + r;
            float pos  = row[grow & (BD - 1)];   // positive column = row mod 512
            float lse  = mx * INV_TEMP + __logf(se);
            partial += lse - pos * INV_TEMP;
        }
    }
    if (lane == 0)
        atomicAdd(out, partial * (1.0f / PB));
}

extern "C" void kernel_launch(void* const* d_in, const int* in_sizes, int n_in,
                              void* d_out, int out_size, void* d_ws, size_t ws_size,
                              hipStream_t stream) {
    const float* feat = (const float*)d_in[0];
    float* out = (float*)d_out;

    float*  zsum = (float*)d_ws;                           // BD*DIM f32   (4 MiB)
    __bf16* fhat = (__bf16*)(zsum + (size_t)BD * DIM);     // PB*DIM bf16  (32 MiB)
    __bf16* zhat = fhat + (size_t)PB * DIM;                // BD*DIM bf16  (2 MiB)

    zsum_kernel<<<(BD * DIM + 255) / 256, 256, 0, stream>>>(feat, zsum, out);
    normalize_rows_kernel<<<PB + BD, 256, 0, stream>>>(feat, zsum, fhat, zhat);
    gemm_lse_kernel<<<PB / 16, 256, 0, stream>>>(fhat, zhat, out);
}
